// Attention_57655640981689
// MI455X (gfx1250) — matrix operand order repaired and checked by reading, not compiled, not run on hardware
//
#include <hip/hip_runtime.h>
#include <stdint.h>

#define BATCH 8
#define SEQ   1024
#define CDIM  512
#define HEADS 8
#define HDIM  64
#define QKV_OUT (3*CDIM)        // 1536
#define ROWS (BATCH*SEQ)        // 8192
#define ATTN_SCALE 0.125f       // 64^-0.5
#define NEG_BIG -1.0e12f

typedef __attribute__((ext_vector_type(16))) __bf16 v16bf;
typedef __attribute__((ext_vector_type(8)))  float  v8f;
typedef unsigned int v4u  __attribute__((ext_vector_type(4)));
typedef int          v8i_ __attribute__((ext_vector_type(8)));
typedef int          v4i_ __attribute__((ext_vector_type(4)));

static __device__ __forceinline__ v8f wmma_bf16(v16bf a, v16bf b, v8f c) {
  return __builtin_amdgcn_wmma_f32_16x16x32_bf16(false, a, false, b, (short)0, c, false, false);
}

#if __has_builtin(__builtin_amdgcn_tensor_load_to_lds) && __has_builtin(__builtin_amdgcn_s_wait_tensorcnt)
#define USE_TDM 1
#endif

// ---------------------------------------------------------------- convert
__global__ void k_convert(const float* __restrict__ x, const float* __restrict__ qw,
                          const float* __restrict__ pw, __bf16* __restrict__ xb,
                          __bf16* __restrict__ wqb, __bf16* __restrict__ pwb) {
  const int n0 = ROWS*CDIM, n1 = QKV_OUT*CDIM, n2 = CDIM*CDIM;
  int i = blockIdx.x * blockDim.x + threadIdx.x;
  if (i < n0)            xb[i]        = (__bf16)x[i];
  else if (i < n0+n1)    wqb[i-n0]    = (__bf16)qw[i-n0];
  else if (i < n0+n1+n2) pwb[i-n0-n1] = (__bf16)pw[i-n0-n1];
}

// ---------------------------------------------------------------- mask gen (JAX threefry2x32, key(42))
static __device__ __forceinline__ uint32_t rotl32(uint32_t v, int r) {
  return (v << r) | (v >> (32 - r));
}
static __device__ __forceinline__ void tf_group(uint32_t& x0, uint32_t& x1,
                                                int r0, int r1, int r2, int r3) {
  x0 += x1; x1 = rotl32(x1, r0); x1 ^= x0;
  x0 += x1; x1 = rotl32(x1, r1); x1 ^= x0;
  x0 += x1; x1 = rotl32(x1, r2); x1 ^= x0;
  x0 += x1; x1 = rotl32(x1, r3); x1 ^= x0;
}

__global__ void k_mask(uint32_t* __restrict__ mw) {
  const uint32_t w = blockIdx.x * blockDim.x + threadIdx.x;
  if (w >= 1048576u) return;                 // words covering flat < 32M
  const uint32_t k0 = 0u, k1 = 42u;
  const uint32_t ks2 = k0 ^ k1 ^ 0x1BD11BDAu;
  uint32_t w0 = 0u, w1 = 0u;
  for (int b = 0; b < 32; ++b) {
    uint32_t x0 = w * 32u + (uint32_t)b;     // counter i
    uint32_t x1 = x0 + 33554432u;            // counter i + half (JAX split-iota pairing)
    x0 += k0; x1 += k1;
    tf_group(x0, x1, 13, 15, 26, 6);  x0 += k1;  x1 += ks2 + 1u;
    tf_group(x0, x1, 17, 29, 16, 24); x0 += ks2; x1 += k0 + 2u;
    tf_group(x0, x1, 13, 15, 26, 6);  x0 += k0;  x1 += k1 + 3u;
    tf_group(x0, x1, 17, 29, 16, 24); x0 += k1;  x1 += ks2 + 4u;
    tf_group(x0, x1, 13, 15, 26, 6);  x0 += ks2; x1 += k0 + 5u;
    // uniform < 0.5  <=>  top bit of random bits == 0
    w0 |= (((~x0) >> 31) & 1u) << b;
    w1 |= (((~x1) >> 31) & 1u) << b;
  }
  mw[w] = w0;
  mw[w + 1048576u] = w1;
}

// ---------------------------------------------------------------- 32x64 register-blocked GEMM core
// acc[mm][g] += A(32xK) x B(Kx64);  A,B row-major bf16 with leading dim K=512.
static __device__ __forceinline__ void gemm_32x64(const __bf16* __restrict__ A,
                                                  const __bf16* __restrict__ B,
                                                  int lane, v8f acc[2][4]) {
  const int mrow = lane & 15;
  const int hi   = lane >> 4;
  const int kb8  = hi * 8, kb16 = hi * 16;
  const __bf16* ar0 = A + (size_t)mrow * CDIM;
  const __bf16* ar1 = A + (size_t)(16 + mrow) * CDIM;
  const __bf16* br0 = B + (size_t)mrow * CDIM;
  const __bf16* br1 = br0 + (size_t)16 * CDIM;
  const __bf16* br2 = br0 + (size_t)32 * CDIM;
  const __bf16* br3 = br0 + (size_t)48 * CDIM;
  #pragma unroll 2
  for (int kc = 0; kc < CDIM; kc += 32) {
    v16bf a0, a1, b0, b1, b2, b3;
    #pragma unroll
    for (int i = 0; i < 8; ++i) {
      a0[i] = ar0[kc + kb16 + i];  a0[8 + i] = ar0[kc + kb16 + 8 + i];
      a1[i] = ar1[kc + kb16 + i];  a1[8 + i] = ar1[kc + kb16 + 8 + i];
    }
    #pragma unroll
    for (int i = 0; i < 16; ++i) {
      b0[i] = br0[kc + kb16 + i];
      b1[i] = br1[kc + kb16 + i];
      b2[i] = br2[kc + kb16 + i];
      b3[i] = br3[kc + kb16 + i];
    }
    acc[0][0] = wmma_bf16(a0, b0, acc[0][0]);
    acc[1][0] = wmma_bf16(a1, b0, acc[1][0]);
    acc[0][1] = wmma_bf16(a0, b1, acc[0][1]);
    acc[1][1] = wmma_bf16(a1, b1, acc[1][1]);
    acc[0][2] = wmma_bf16(a0, b2, acc[0][2]);
    acc[1][2] = wmma_bf16(a1, b2, acc[1][2]);
    acc[0][3] = wmma_bf16(a0, b3, acc[0][3]);
    acc[1][3] = wmma_bf16(a1, b3, acc[1][3]);
  }
}

// ---------------------------------------------------------------- QKV GEMM: [8192,512] x [1536,512]^T
__global__ void __launch_bounds__(256) k_qkv(const __bf16* __restrict__ xb,
                                             const __bf16* __restrict__ wqb,
                                             const float* __restrict__ qbias,
                                             __bf16* __restrict__ Q,
                                             __bf16* __restrict__ K,
                                             __bf16* __restrict__ V) {
  const int NT = QKV_OUT / 64;                       // 24 col tiles
  const int wid = blockIdx.x * 8 + (threadIdx.x >> 5);
  if (wid >= (ROWS/32) * NT) return;
  const int lane = threadIdx.x & 31;
  const int mt = wid / NT, nt = wid % NT;
  const int m0 = mt * 32, n0 = nt * 64;
  const int mrow = lane & 15;
  const int hi   = lane >> 4;

  v8f acc[2][4];
  #pragma unroll
  for (int mm = 0; mm < 2; ++mm)
    #pragma unroll
    for (int g = 0; g < 4; ++g)
      acc[mm][g] = (v8f){0.f,0.f,0.f,0.f,0.f,0.f,0.f,0.f};

  gemm_32x64(xb + (size_t)m0 * CDIM, wqb + (size_t)n0 * CDIM, lane, acc);

  // 64-wide tile stays within one (q|k|v, head) pair
  const int which = n0 / CDIM;
  const int h = (n0 % CDIM) / HDIM;
  __bf16* dst = (which == 0) ? Q : (which == 1) ? K : V;
  #pragma unroll
  for (int g = 0; g < 4; ++g) {
    const int d = g * 16 + mrow;
    const float bias = qbias[n0 + d];
    #pragma unroll
    for (int mm = 0; mm < 2; ++mm)
      #pragma unroll
      for (int r = 0; r < 8; ++r) {
        const int row  = m0 + mm * 16 + r + 8 * hi;
        const int bidx = row / SEQ, nrow = row % SEQ;
        dst[((size_t)(bidx * HEADS + h) * SEQ + nrow) * HDIM + d] =
            (__bf16)(acc[mm][g][r] + bias);
      }
  }
}

// ---------------------------------------------------------------- flash attention (1 wave / 32 q-rows)
__global__ void __launch_bounds__(32) k_attn(const __bf16* __restrict__ Q,
                                             const __bf16* __restrict__ Km,
                                             const __bf16* __restrict__ Vm,
                                             const uint32_t* __restrict__ maskw,
                                             __bf16* __restrict__ aO) {
  __shared__ __bf16 Plds[2][16 * 34];  // two P tiles 16x32, row stride 34 halves
  __shared__ __bf16 Vlds[32 * 72];     // V tile 32x64, row stride 72 halves (144B)

  const int lane  = threadIdx.x & 31;
  const int bh    = blockIdx.x >> 5;            // 32 q-blocks (of 32 rows) per (b,h)
  const int qbase = (blockIdx.x & 31) * 32;
  const int mrow  = lane & 15;
  const int hi    = lane >> 4;
  const int kb8   = hi * 8, kb16 = hi * 16;

  // Q A-matrices for both 16-row tiles (head-dim chunks 0..31, 32..63)
  v16bf qa[2][2];
  #pragma unroll
  for (int t = 0; t < 2; ++t) {
    const __bf16* qrow = Q + ((size_t)bh * SEQ + qbase + t * 16 + mrow) * HDIM;
    #pragma unroll
    for (int i = 0; i < 8; ++i) {
      qa[t][0][i]     = qrow[kb16 + i];       qa[t][0][8 + i] = qrow[8 + kb16 + i];
      qa[t][1][i]     = qrow[32 + kb16 + i];  qa[t][1][8 + i] = qrow[40 + kb16 + i];
    }
  }

  v8f o[2][4];
  float mrun[2][8], lrun[2][8];
  #pragma unroll
  for (int t = 0; t < 2; ++t)
    #pragma unroll
    for (int r = 0; r < 8; ++r) {
      mrun[t][r] = -3.0e38f; lrun[t][r] = 0.f;
      if (r < 4) o[t][r] = (v8f){0.f,0.f,0.f,0.f,0.f,0.f,0.f,0.f};
    }

  const uint32_t vlds_off = (uint32_t)(uintptr_t)(&Vlds[0]);   // wave-relative LDS byte offset

  for (int j = 0; j < SEQ; j += 32) {
    __syncthreads();
    // ---- stage V rows j..j+31 into LDS (padded rows of 72 halves) ----
#if defined(USE_TDM)
    {
      // Tensor DMA: 2D tile 32x64 of 2B elements, HW pads +16B per 128B row.
      const uint64_t ga = (uint64_t)(uintptr_t)(Vm + ((size_t)bh * SEQ + j) * HDIM);
      v4u g0;
      g0[0] = 1u;                                              // count=1 (valid D#)
      g0[1] = vlds_off;                                        // lds_addr
      g0[2] = (uint32_t)ga;                                    // global_addr[31:0]
      g0[3] = ((uint32_t)(ga >> 32) & 0x01FFFFFFu) | (2u << 30); // addr[56:32] | type=2
      v8i_ g1;
      g1[0] = (int)((1u << 16) | (1u << 20) | (4u << 22) | (3u << 25));
              // data_size=2B | pad_enable | pad_interval=32dw(128B) | pad_amount=4dw(16B)
      g1[1] = (int)(64u   << 16);   // tensor_dim0 = 64 (bits 79:48, low half)
      g1[2] = (int)(1024u << 16);   // tensor_dim0 hi=0 | tensor_dim1 = 1024 (low half)
      g1[3] = (int)(64u   << 16);   // tensor_dim1 hi=0 | tile_dim0 = 64
      g1[4] = 32;                   // tile_dim1 = 32, tile_dim2 = 0
      g1[5] = 64;                   // tensor_dim0_stride = 64 (row stride, elements)
      g1[6] = 0;                    // stride hi | tensor_dim1_stride lo
      g1[7] = 0;
      const v4i_ gz = {0, 0, 0, 0};
#if __clang_major__ >= 23
      const v8i_ gz8 = {0, 0, 0, 0, 0, 0, 0, 0};
      __builtin_amdgcn_tensor_load_to_lds(g0, g1, gz, gz, gz8, 0);
#else
      __builtin_amdgcn_tensor_load_to_lds(g0, g1, gz, gz, 0);
#endif
    }
#else
    {
      const ulonglong2* vsrc = (const ulonglong2*)(Vm + ((size_t)bh * SEQ + j + lane) * HDIM);
      ulonglong2* vdst = (ulonglong2*)(&Vlds[lane * 72]);
      #pragma unroll
      for (int c = 0; c < 8; ++c) vdst[c] = vsrc[c];
    }
#endif
    // ---- S = Q K^T : K B-operands loaded once, reused by both q-tiles ----
    const __bf16* krow0 = Km + ((size_t)bh * SEQ + j + mrow) * HDIM;  // keys j + n
    const __bf16* krow1 = krow0 + 16 * HDIM;                          // keys j+16 + n
    v16bf b00, b01, b10, b11;
    #pragma unroll
    for (int i = 0; i < 16; ++i) {
      b00[i] = krow0[kb16 + i];
      b01[i] = krow0[32 + kb16 + i];
      b10[i] = krow1[kb16 + i];
      b11[i] = krow1[32 + kb16 + i];
    }
    #pragma unroll
    for (int t = 0; t < 2; ++t) {
      v8f s0v = {0.f,0.f,0.f,0.f,0.f,0.f,0.f,0.f};
      v8f s1v = s0v;
      s0v = wmma_bf16(qa[t][0], b00, s0v);
      s0v = wmma_bf16(qa[t][1], b01, s0v);
      s1v = wmma_bf16(qa[t][0], b10, s1v);
      s1v = wmma_bf16(qa[t][1], b11, s1v);
      // mask + online softmax (row lives in one 16-lane half; xor 1/2/4/8 stays inside)
      #pragma unroll
      for (int r = 0; r < 8; ++r) {
        const int qrow = qbase + t * 16 + r + 8 * hi;
        const uint32_t mwrd = maskw[((uint32_t)bh * SEQ + (uint32_t)qrow) * 32u + (uint32_t)(j >> 5)];
        float s0 = s0v[r] * ATTN_SCALE + (((mwrd >> mrow)        & 1u) ? NEG_BIG : 0.f);
        float s1 = s1v[r] * ATTN_SCALE + (((mwrd >> (16 + mrow)) & 1u) ? NEG_BIG : 0.f);
        float mx = fmaxf(s0, s1);
        mx = fmaxf(mx, __shfl_xor(mx, 1, 32));
        mx = fmaxf(mx, __shfl_xor(mx, 2, 32));
        mx = fmaxf(mx, __shfl_xor(mx, 4, 32));
        mx = fmaxf(mx, __shfl_xor(mx, 8, 32));
        const float mnew = fmaxf(mrun[t][r], mx);
        const float corr = __expf(mrun[t][r] - mnew);
        const float e0 = __expf(s0 - mnew);
        const float e1 = __expf(s1 - mnew);
        float rs = e0 + e1;
        rs += __shfl_xor(rs, 1, 32);
        rs += __shfl_xor(rs, 2, 32);
        rs += __shfl_xor(rs, 4, 32);
        rs += __shfl_xor(rs, 8, 32);
        lrun[t][r] = lrun[t][r] * corr + rs;
        mrun[t][r] = mnew;
        o[t][0][r] *= corr; o[t][1][r] *= corr; o[t][2][r] *= corr; o[t][3][r] *= corr;
        Plds[t][(r + 8 * hi) * 34 + mrow]      = (__bf16)e0;   // D-layout -> row-major P
        Plds[t][(r + 8 * hi) * 34 + 16 + mrow] = (__bf16)e1;
      }
    }
#if defined(USE_TDM)
    __builtin_amdgcn_s_wait_tensorcnt(0);
#endif
    __syncthreads();
    // ---- O += P @ V : A = P per tile, B = V per dout block (shared) ----
    v16bf ap[2];
    #pragma unroll
    for (int t = 0; t < 2; ++t)
      #pragma unroll
      for (int i = 0; i < 8; ++i) {
        ap[t][i]     = Plds[t][mrow * 34 + kb16 + i];
        ap[t][8 + i] = Plds[t][mrow * 34 + kb16 + 8 + i];
      }
    #pragma unroll
    for (int d = 0; d < 4; ++d) {
      v16bf bv;
      #pragma unroll
      for (int i = 0; i < 16; ++i) bv[i] = Vlds[(kb16 + i) * 72 + d * 16 + mrow];
      o[0][d] = wmma_bf16(ap[0], bv, o[0][d]);
      o[1][d] = wmma_bf16(ap[1], bv, o[1][d]);
    }
  }
  // epilogue: normalize, write [B, N, H*64] bf16 for proj GEMM
  const int b = bh >> 3, h = bh & 7;
  #pragma unroll
  for (int t = 0; t < 2; ++t)
    #pragma unroll
    for (int r = 0; r < 8; ++r) {
      const float inv = 1.f / lrun[t][r];
      const int row = qbase + t * 16 + r + 8 * hi;
      const size_t base = ((size_t)b * SEQ + row) * CDIM + h * HDIM + mrow;
      aO[base +  0] = (__bf16)(o[t][0][r] * inv);
      aO[base + 16] = (__bf16)(o[t][1][r] * inv);
      aO[base + 32] = (__bf16)(o[t][2][r] * inv);
      aO[base + 48] = (__bf16)(o[t][3][r] * inv);
    }
}

// ---------------------------------------------------------------- projection GEMM -> f32 output
__global__ void __launch_bounds__(256) k_proj(const __bf16* __restrict__ aO,
                                              const __bf16* __restrict__ pwb,
                                              const float* __restrict__ pbias,
                                              float* __restrict__ out) {
  const int NT = CDIM / 64;                           // 8 col tiles
  const int wid = blockIdx.x * 8 + (threadIdx.x >> 5);
  if (wid >= (ROWS/32) * NT) return;
  const int lane = threadIdx.x & 31;
  const int mt = wid / NT, nt = wid % NT;
  const int m0 = mt * 32, n0 = nt * 64;
  const int mrow = lane & 15;
  const int hi   = lane >> 4;

  v8f acc[2][4];
  #pragma unroll
  for (int mm = 0; mm < 2; ++mm)
    #pragma unroll
    for (int g = 0; g < 4; ++g)
      acc[mm][g] = (v8f){0.f,0.f,0.f,0.f,0.f,0.f,0.f,0.f};

  gemm_32x64(aO + (size_t)m0 * CDIM, pwb + (size_t)n0 * CDIM, lane, acc);

  #pragma unroll
  for (int g = 0; g < 4; ++g) {
    const int ncol = n0 + g * 16 + mrow;
    const float bias = pbias[ncol];
    #pragma unroll
    for (int mm = 0; mm < 2; ++mm)
      #pragma unroll
      for (int r = 0; r < 8; ++r)
        out[(size_t)(m0 + mm * 16 + r + 8 * hi) * CDIM + ncol] = acc[mm][g][r] + bias;
  }
}

// ---------------------------------------------------------------- launch
extern "C" void kernel_launch(void* const* d_in, const int* in_sizes, int n_in,
                              void* d_out, int out_size, void* d_ws, size_t ws_size,
                              hipStream_t stream) {
  const float* x  = (const float*)d_in[0];
  const float* qw = (const float*)d_in[1];
  const float* qb = (const float*)d_in[2];
  const float* pw = (const float*)d_in[3];
  const float* pb = (const float*)d_in[4];

  __bf16* ws  = (__bf16*)d_ws;
  __bf16* xb  = ws;                           // 4,194,304 bf16
  __bf16* wqb = xb  + (size_t)ROWS * CDIM;    // 786,432
  __bf16* pwb = wqb + (size_t)QKV_OUT * CDIM; // 262,144
  __bf16* Qb  = pwb + (size_t)CDIM * CDIM;    // 4,194,304 each
  __bf16* Kb  = Qb  + (size_t)BATCH * HEADS * SEQ * HDIM;
  __bf16* Vb  = Kb  + (size_t)BATCH * HEADS * SEQ * HDIM;
  __bf16* aO  = Vb  + (size_t)BATCH * HEADS * SEQ * HDIM;
  uint32_t* mw = (uint32_t*)(aO + (size_t)ROWS * CDIM);  // 2,097,152 words (8MB)

  const int nconv = ROWS*CDIM + QKV_OUT*CDIM + CDIM*CDIM;
  k_convert<<<(nconv + 255) / 256, 256, 0, stream>>>(x, qw, pw, xb, wqb, pwb);
  k_mask<<<1048576 / 256, 256, 0, stream>>>(mw);
  k_qkv<<<(ROWS/32) * (QKV_OUT/64) / 8, 256, 0, stream>>>(xb, wqb, qb, Qb, Kb, Vb);
  k_attn<<<BATCH * HEADS * (SEQ / 32), 32, 0, stream>>>(Qb, Kb, Vb, mw, aO);
  k_proj<<<(ROWS/32) * (CDIM/64) / 8, 256, 0, stream>>>(aO, pwb, pb, (float*)d_out);
}
